// LinearAttentionBlock_54322746359986
// MI455X (gfx1250) — matrix operand order repaired and checked by reading, not compiled, not run on hardware
//
#include <hip/hip_runtime.h>
#include <hip/hip_bf16.h>

// ---------------------------------------------------------------------------
// LinearAttentionBlock fused pipeline for MI455X (gfx1250, wave32, WMMA).
// Shapes: b=8, n=4096, dim=512, heads=8, head_size=64, hidden=512.
// GEMMs: v_wmma_f32_16x16x32_f16 (f16 operands, f32 accumulate).
// context GEMM: split-K + LDS staging via async global->LDS copies (ASYNCcnt).
// ---------------------------------------------------------------------------

typedef __attribute__((ext_vector_type(16))) _Float16 v16h;
typedef __attribute__((ext_vector_type(8)))  float    v8f;
typedef int v2i __attribute__((vector_size(2 * sizeof(int))));

#define B_SZ    8
#define N_SEQ   4096
#define DIM     512
#define HEADS   8
#define HSZ     64
#define QKVW    1536            // 3 * hidden
#define M_TOT   (B_SZ * N_SEQ)  // 32768
#define KCHUNK  512             // split-K depth for context GEMM
#define ROWP    68              // padded LDS row (elements), keeps 8B align

__device__ __forceinline__ v8f wmma_f16(v16h a, v16h b, v8f c) {
    // (neg_a, A, neg_b, B, c_mod, C, reuse_a, reuse_b)
    return __builtin_amdgcn_wmma_f32_16x16x32_f16(false, a, false, b,
                                                  (short)0, c, false, false);
}

// ---- async global -> LDS copy (8 bytes / lane), gfx1250 ASYNCcnt path -----
__device__ __forceinline__ void async_copy_b64(const void* g, void* l) {
#if __has_builtin(__builtin_amdgcn_global_load_async_to_lds_b64)
    __builtin_amdgcn_global_load_async_to_lds_b64(
        (__attribute__((address_space(1))) v2i*)(v2i*)const_cast<void*>(g),
        (__attribute__((address_space(3))) v2i*)l, 0, 0);
#else
    *(unsigned long long*)l = *(const unsigned long long*)g;
#endif
}
__device__ __forceinline__ void async_wait_all() {
#if __has_builtin(__builtin_amdgcn_global_load_async_to_lds_b64)
  #if __has_builtin(__builtin_amdgcn_s_wait_asynccnt)
    __builtin_amdgcn_s_wait_asynccnt(0);
  #else
    asm volatile("s_wait_asynccnt 0" ::: "memory");
  #endif
#endif
}

// ---- fragment loaders (contiguous k order: k = 16*half + s) ----------------------------
// A-fragment (16x32, MxK) from row-major f32.
__device__ __forceinline__ v16h load_a_f32(const float* __restrict__ A, int lda,
                                           int k0, int lane) {
    const int m = lane & 15, half = lane >> 4;
    const float* p = A + (size_t)m * lda + k0 + half * 16;
    v16h a;
#pragma unroll
    for (int i = 0; i < 8; ++i) {
        a[i]     = (_Float16)p[i];        // K = k0 + half*16 + i
        a[8 + i] = (_Float16)p[8 + i];    // K = k0 + half*16 + 8 + i
    }
    return a;
}

// A-fragment from row-major f16.
__device__ __forceinline__ v16h load_a_f16(const _Float16* __restrict__ A, int lda,
                                           int k0, int lane) {
    const int m = lane & 15, half = lane >> 4;
    const _Float16* p = A + (size_t)m * lda + k0 + half * 16;
    v16h a;
#pragma unroll
    for (int i = 0; i < 8; ++i) { a[i] = p[i]; a[8 + i] = p[8 + i]; }
    return a;
}

// B-fragment (32x16, KxN) where B[k][n] = W[n][k], W row-major f16.
// 16 contiguous f16 (32B, aligned) per lane.
__device__ __forceinline__ v16h load_bT_f16(const _Float16* __restrict__ W, int ldw,
                                            int n0, int k0, int lane) {
    const int n = lane & 15, half = lane >> 4;
    return *(const v16h*)(W + (size_t)(n0 + n) * ldw + k0 + half * 16);
}

// ---------------------------------------------------------------------------
// K0: f32 -> f16 weight conversion (run once per launch for w_qkv, w_out).
// ---------------------------------------------------------------------------
__global__ void __launch_bounds__(256)
cvt_f32_to_f16(const float* __restrict__ src, _Float16* __restrict__ dst, int n) {
    const int i = blockIdx.x * 256 + threadIdx.x;
    if (i < n) dst[i] = (_Float16)src[i];
}

// ---------------------------------------------------------------------------
// K1: qkv = x @ w_qkv^T   (M=32768, N=1536, K=512), output f16.
// grid (2048, 3) x 256 thr; wave owns a 16x64 tile.
// ---------------------------------------------------------------------------
__global__ void __launch_bounds__(256)
qkv_gemm(const float* __restrict__ x, const _Float16* __restrict__ wqkv_h,
         _Float16* __restrict__ qkv) {
    const int lane = threadIdx.x & 31;
    const int wv   = threadIdx.x >> 5;
    const int m0   = blockIdx.x * 16;
    const int n0   = (blockIdx.y * 8 + wv) * 64;
    const float* Arow = x + (size_t)m0 * DIM;

    v8f acc[4] = {v8f{}, v8f{}, v8f{}, v8f{}};
    for (int k0 = 0; k0 < DIM; k0 += 32) {
        if (k0 + 32 < DIM)  // pull next A slab toward the caches
            __builtin_prefetch(Arow + (size_t)(lane & 15) * DIM + k0 + 32, 0, 3);
        v16h a = load_a_f32(Arow, DIM, k0, lane);
#pragma unroll
        for (int t = 0; t < 4; ++t) {
            v16h b = load_bT_f16(wqkv_h, DIM, n0 + t * 16, k0, lane);
            acc[t] = wmma_f16(a, b, acc[t]);
        }
    }
    const int nn = lane & 15, half = lane >> 4;
#pragma unroll
    for (int t = 0; t < 4; ++t)
#pragma unroll
        for (int r = 0; r < 8; ++r)
            qkv[(size_t)(m0 + half * 8 + r) * QKVW + n0 + t * 16 + nn] =
                (_Float16)acc[t][r];
}

// ---------------------------------------------------------------------------
// K2a: k-softmax partial stats (online-softmax chunks), coalesced reads.
// grid (b=8, h=8, c=8) x 256 thr; chunk = 512 sequence rows.
// ---------------------------------------------------------------------------
__global__ void __launch_bounds__(256)
ksm_partial(const _Float16* __restrict__ qkv,
            float* __restrict__ pmax, float* __restrict__ psum) {
    const int b = blockIdx.x, h = blockIdx.y, c = blockIdx.z;
    const int d = threadIdx.x & 63, rg = threadIdx.x >> 6;   // 4 row groups
    const _Float16* base =
        qkv + ((size_t)b * N_SEQ + c * 512) * QKVW + DIM + h * HSZ + d;
    __shared__ float smax[4][64], ssum[4][64], lmax[64];

    float mx = -1e30f;
    for (int n = rg; n < 512; n += 4)
        mx = fmaxf(mx, (float)base[(size_t)n * QKVW]);
    smax[rg][d] = mx;
    __syncthreads();
    if (rg == 0)
        lmax[d] = fmaxf(fmaxf(smax[0][d], smax[1][d]),
                        fmaxf(smax[2][d], smax[3][d]));
    __syncthreads();
    const float m2 = lmax[d];

    float sm = 0.f;
    for (int n = rg; n < 512; n += 4)
        sm += __expf((float)base[(size_t)n * QKVW] - m2);
    ssum[rg][d] = sm;
    __syncthreads();
    if (rg == 0) {
        const int idx = (((b * HEADS) + h) * 8 + c) * 64 + d;
        pmax[idx] = m2;
        psum[idx] = ssum[0][d] + ssum[1][d] + ssum[2][d] + ssum[3][d];
    }
}

// K2b: merge the 8 chunk (max,sum) pairs per (b, h*64+d).
__global__ void __launch_bounds__(256)
ksm_combine(const float* __restrict__ pmax, const float* __restrict__ psum,
            float* __restrict__ kmax, float* __restrict__ ksum) {
    const int idx = blockIdx.x * 256 + threadIdx.x;   // < 4096
    if (idx >= B_SZ * DIM) return;
    const int b = idx >> 9, j = idx & 511;
    const int h = j >> 6, d = j & 63;
    const float* pm = pmax + ((b * HEADS + h) * 8) * 64 + d;
    const float* ps = psum + ((b * HEADS + h) * 8) * 64 + d;
    float g = -1e30f;
#pragma unroll
    for (int c = 0; c < 8; ++c) g = fmaxf(g, pm[c * 64]);
    float s = 0.f;
#pragma unroll
    for (int c = 0; c < 8; ++c) s += ps[c * 64] * __expf(pm[c * 64] - g);
    kmax[idx] = g;
    ksum[idx] = s;
}

// ---------------------------------------------------------------------------
// K3: q-softmax stats over head_size=64. One block per row m; wave -> head.
// ---------------------------------------------------------------------------
__global__ void __launch_bounds__(256)
qsm_stats(const _Float16* __restrict__ qkv,
          float* __restrict__ qmax, float* __restrict__ qsum) {
    const int m = blockIdx.x;
    const int lane = threadIdx.x & 31;
    const int h = threadIdx.x >> 5;
    const _Float16* row = qkv + (size_t)m * QKVW + h * HSZ;
    float v0 = (float)row[lane], v1 = (float)row[lane + 32];
    float mx = fmaxf(v0, v1);
#pragma unroll
    for (int off = 16; off > 0; off >>= 1)
        mx = fmaxf(mx, __shfl_xor(mx, off, 32));
    float sm = __expf(v0 - mx) + __expf(v1 - mx);
#pragma unroll
    for (int off = 16; off > 0; off >>= 1)
        sm += __shfl_xor(sm, off, 32);
    if (lane == 0) { qmax[m * HEADS + h] = mx; qsum[m * HEADS + h] = sm; }
}

// ---------------------------------------------------------------------------
// K4: context accumulation, split-K.
// ctxacc[b,h,d,e] += sum_{n in chunk} exp(k[n,d]-kmax[d]) * v[n,e]
// grid (h=8, b=8, c=8) x 512 thr (16 waves; wave -> one 16x16 tile).
// 32x64 k/v tiles staged in LDS via async global->LDS copies (coalesced).
// ---------------------------------------------------------------------------
__global__ void __launch_bounds__(512)
context_gemm(const _Float16* __restrict__ qkv, const float* __restrict__ kmax,
             float* __restrict__ ctxacc) {
    __shared__ _Float16 Ks[32 * ROWP];
    __shared__ _Float16 Vs[32 * ROWP];
    const int h = blockIdx.x, b = blockIdx.y, c = blockIdx.z;
    const int tid  = threadIdx.x;
    const int lane = tid & 31, wv = tid >> 5;
    const int d0 = (wv >> 2) * 16, e0 = (wv & 3) * 16;
    const int mrow = lane & 15, half = lane >> 4;
    const int dloc = d0 + mrow;                      // A-matrix row (d)
    const int eloc = e0 + mrow;                      // B-matrix col (e)
    const float mxa = kmax[b * DIM + h * HSZ + dloc];

    // staging: each of 512 threads moves one 8B group per tile
    const int sn = tid >> 4;                         // 0..31 (tile row n)
    const int sd = (tid & 15) * 4;                   // 0..60 (d/e group)
    const size_t rowbase = ((size_t)b * N_SEQ + (size_t)c * KCHUNK) * QKVW + h * HSZ;
    const _Float16* kg = qkv + rowbase + DIM     + sd;
    const _Float16* vg = qkv + rowbase + 2 * DIM + sd;

    v8f acc = v8f{};
    for (int kk = 0; kk < KCHUNK; kk += 32) {
        __syncthreads();                             // LDS free for refill
        async_copy_b64(kg + (size_t)(kk + sn) * QKVW, &Ks[sn * ROWP + sd]);
        async_copy_b64(vg + (size_t)(kk + sn) * QKVW, &Vs[sn * ROWP + sd]);
        async_wait_all();
        __syncthreads();                             // tiles visible to all waves

        v16h a, bf;
#pragma unroll
        for (int i = 0; i < 8; ++i) {
            a[i]     = (_Float16)__expf((float)Ks[(half * 16 + i) * ROWP + dloc] - mxa);
            a[8 + i] = (_Float16)__expf((float)Ks[(half * 16 + i + 8) * ROWP + dloc] - mxa);
        }
#pragma unroll
        for (int i = 0; i < 16; ++i)
            bf[i] = Vs[(half * 16 + i) * ROWP + eloc];
        acc = wmma_f16(a, bf, acc);
    }
#pragma unroll
    for (int r = 0; r < 8; ++r) {
        const int d = d0 + half * 8 + r;
        atomicAdd(&ctxacc[(((size_t)b * HEADS + h) * HSZ + d) * HSZ + eloc], acc[r]);
    }
}

// K4b: ctx = ctxacc / ksum (row-wise 1/sum of the k softmax).
__global__ void __launch_bounds__(256)
ctx_finish(const float* __restrict__ ctxacc, const float* __restrict__ ksum,
           float* __restrict__ ctx) {
    const int idx = blockIdx.x * 256 + threadIdx.x;  // < 262144
    const int d = (idx >> 6) & 63, h = (idx >> 12) & 7, b = idx >> 15;
    ctx[idx] = ctxacc[idx] / ksum[b * DIM + h * HSZ + d];
}

// ---------------------------------------------------------------------------
// K5: out2[b,n,h*64+e] = (scale/qsum) * sum_d exp(q-qmax)[n,d] * ctx[d,e]
// Per (b,h): M=4096, N=64, K=64. grid (32,8,8) x 256 thr; wave -> 16x64 tile.
// ---------------------------------------------------------------------------
__global__ void __launch_bounds__(256)
attn_out_gemm(const _Float16* __restrict__ qkv, const float* __restrict__ ctx,
              const float* __restrict__ qmax, const float* __restrict__ qsum,
              _Float16* __restrict__ out2) {
    const int b = blockIdx.y, h = blockIdx.z;
    const int lane = threadIdx.x & 31, wv = threadIdx.x >> 5;
    const int m0 = (blockIdx.x * 8 + wv) * 16;
    const size_t mbase = (size_t)b * N_SEQ + m0;
    const int mrow = lane & 15, half = lane >> 4;

    const _Float16* qrow = qkv + mbase * QKVW + h * HSZ;
    const float* cbase   = ctx + ((size_t)b * HEADS + h) * HSZ * HSZ;
    const float mxa = qmax[(mbase + mrow) * HEADS + h];

    v8f acc[4] = {v8f{}, v8f{}, v8f{}, v8f{}};
    for (int k0 = 0; k0 < HSZ; k0 += 32) {
        v16h a;
#pragma unroll
        for (int i = 0; i < 8; ++i) {
            a[i]     = (_Float16)__expf((float)qrow[(size_t)mrow * QKVW + k0 + half * 16 + i] - mxa);
            a[8 + i] = (_Float16)__expf((float)qrow[(size_t)mrow * QKVW + k0 + 8 + half * 16 + i] - mxa);
        }
#pragma unroll
        for (int t = 0; t < 4; ++t) {
            v16h bf;
#pragma unroll
            for (int i = 0; i < 16; ++i)
                bf[i] = (_Float16)cbase[(k0 + half * 16 + i) * HSZ + t * 16 + mrow];
            acc[t] = wmma_f16(a, bf, acc[t]);
        }
    }
    const float scale = 0.125f;                      // 64^-0.5
#pragma unroll
    for (int r = 0; r < 8; ++r) {
        const int m = half * 8 + r;
        const float rs = scale / qsum[(mbase + m) * HEADS + h];
#pragma unroll
        for (int t = 0; t < 4; ++t)
            out2[(mbase + m) * DIM + h * HSZ + t * 16 + mrow] =
                (_Float16)(acc[t][r] * rs);
    }
}

// ---------------------------------------------------------------------------
// K6: out = LayerNorm_gamma(out2 @ w_out^T). M=32768, N=K=512.
// Workgroup = full 16x512 row tile (8 waves x 16x64) so LN fuses in epilogue.
// ---------------------------------------------------------------------------
__global__ void __launch_bounds__(256)
out_proj_ln(const _Float16* __restrict__ out2, const _Float16* __restrict__ wout_h,
            const float* __restrict__ gamma, float* __restrict__ out) {
    __shared__ float tile[16 * 512];
    __shared__ float s_mu[16], s_rs[16];

    const int lane = threadIdx.x & 31, wv = threadIdx.x >> 5;
    const int m0 = blockIdx.x * 16;
    const int n0 = wv * 64;
    const _Float16* Arow = out2 + (size_t)m0 * DIM;

    v8f acc[4] = {v8f{}, v8f{}, v8f{}, v8f{}};
    for (int k0 = 0; k0 < DIM; k0 += 32) {
        v16h a = load_a_f16(Arow, DIM, k0, lane);
#pragma unroll
        for (int t = 0; t < 4; ++t) {
            v16h b = load_bT_f16(wout_h, DIM, n0 + t * 16, k0, lane);
            acc[t] = wmma_f16(a, b, acc[t]);
        }
    }
    const int nn = lane & 15, half = lane >> 4;
#pragma unroll
    for (int t = 0; t < 4; ++t)
#pragma unroll
        for (int r = 0; r < 8; ++r)
            tile[(half * 8 + r) * 512 + n0 + t * 16 + nn] = acc[t][r];
    __syncthreads();

#pragma unroll
    for (int rr = 0; rr < 2; ++rr) {                 // wave wv -> rows 2wv(+1)
        const int row = wv * 2 + rr;
        float s = 0.f, s2 = 0.f;
        for (int j = lane; j < 512; j += 32) {
            const float v = tile[row * 512 + j];
            s += v; s2 += v * v;
        }
#pragma unroll
        for (int off = 16; off > 0; off >>= 1) {
            s  += __shfl_xor(s,  off, 32);
            s2 += __shfl_xor(s2, off, 32);
        }
        if (lane == 0) {
            const float mu  = s * (1.f / 512.f);
            const float var = s2 * (1.f / 512.f) - mu * mu;
            s_mu[row] = mu;
            s_rs[row] = rsqrtf(var + 1e-5f);
        }
    }
    __syncthreads();

    for (int idx = threadIdx.x; idx < 16 * 512; idx += 256) {
        const int r = idx >> 9, j = idx & 511;
        out[(size_t)(m0 + r) * DIM + j] = (tile[idx] - s_mu[r]) * s_rs[r] * gamma[j];
    }
}

// ---------------------------------------------------------------------------
extern "C" void kernel_launch(void* const* d_in, const int* in_sizes, int n_in,
                              void* d_out, int out_size, void* d_ws, size_t ws_size,
                              hipStream_t stream) {
    const float* x     = (const float*)d_in[0];   // [8,4096,512]
    const float* wqkv  = (const float*)d_in[1];   // [1536,512]
    const float* wout  = (const float*)d_in[2];   // [512,512]
    const float* gamma = (const float*)d_in[3];   // [512]

    char* ws = (char*)d_ws;
    size_t off = 0;
    _Float16* qkv    = (_Float16*)(ws + off); off += (size_t)M_TOT * QKVW * sizeof(_Float16); // 96 MB
    _Float16* wqkv_h = (_Float16*)(ws + off); off += (size_t)QKVW * DIM * sizeof(_Float16);
    _Float16* wout_h = (_Float16*)(ws + off); off += (size_t)DIM * DIM * sizeof(_Float16);
    float*    pmax   = (float*)(ws + off);    off += (size_t)B_SZ * HEADS * 8 * HSZ * sizeof(float);
    float*    psum   = (float*)(ws + off);    off += (size_t)B_SZ * HEADS * 8 * HSZ * sizeof(float);
    float*    kmax   = (float*)(ws + off);    off += (size_t)B_SZ * DIM * sizeof(float);
    float*    ksum   = (float*)(ws + off);    off += (size_t)B_SZ * DIM * sizeof(float);
    float*    qmax   = (float*)(ws + off);    off += (size_t)M_TOT * HEADS * sizeof(float);
    float*    qsum   = (float*)(ws + off);    off += (size_t)M_TOT * HEADS * sizeof(float);
    float*    ctxacc = (float*)(ws + off);    off += (size_t)B_SZ * HEADS * HSZ * HSZ * sizeof(float);
    float*    ctx    = (float*)(ws + off);    off += (size_t)B_SZ * HEADS * HSZ * HSZ * sizeof(float);
    _Float16* out2   = (_Float16*)(ws + off); off += (size_t)M_TOT * DIM * sizeof(_Float16);  // 32 MB

    cvt_f32_to_f16<<<(QKVW * DIM + 255) / 256, 256, 0, stream>>>(wqkv, wqkv_h, QKVW * DIM);
    cvt_f32_to_f16<<<(DIM * DIM + 255) / 256, 256, 0, stream>>>(wout, wout_h, DIM * DIM);

    qkv_gemm   <<<dim3(M_TOT / 16, 3), 256, 0, stream>>>(x, wqkv_h, qkv);
    ksm_partial<<<dim3(B_SZ, HEADS, 8), 256, 0, stream>>>(qkv, pmax, psum);
    ksm_combine<<<(B_SZ * DIM + 255) / 256, 256, 0, stream>>>(pmax, psum, kmax, ksum);
    qsm_stats  <<<M_TOT, 256, 0, stream>>>(qkv, qmax, qsum);

    (void)hipMemsetAsync(ctxacc, 0,
                         (size_t)B_SZ * HEADS * HSZ * HSZ * sizeof(float), stream);
    context_gemm<<<dim3(HEADS, B_SZ, N_SEQ / KCHUNK), 512, 0, stream>>>(qkv, kmax, ctxacc);
    ctx_finish  <<<(B_SZ * HEADS * HSZ * HSZ) / 256, 256, 0, stream>>>(ctxacc, ksum, ctx);

    attn_out_gemm<<<dim3(32, B_SZ, HEADS), 256, 0, stream>>>(qkv, ctx, qmax, qsum, out2);
    out_proj_ln  <<<M_TOT / 16, 256, 0, stream>>>(out2, wout_h, gamma, (float*)d_out);
}
